// MyLSTM_50105088475181
// MI455X (gfx1250) — matrix operand order repaired and checked by reading, not compiled, not run on hardware
//
#include <hip/hip_runtime.h>

// Problem constants (match reference)
#define BB   32       // batch
#define SS   2048     // sequence length
#define FF   512      // input features (== H for layer 1)
#define HH   512      // hidden size
#define KK   1024     // concatenated K = F + H
#define NBLK 64       // persistent blocks; each owns HB hidden units
#define HB   8        // hidden units per block
#define NL   32       // local gate columns per block (4 gates * HB)
#define TPB  128      // 4 waves of 32
#define LP   1032     // padded LDS K-stride: 2064B rows -> 16B aligned, 516 dwords (banks 4 apart, 16 gather lanes distinct)

typedef __attribute__((ext_vector_type(16))) _Float16 v16h;
typedef __attribute__((ext_vector_type(8)))  float    v8f;

__device__ __forceinline__ float sigmoidf_(float x) { return 1.0f / (1.0f + __expf(-x)); }

// f32 -> f16 weight conversion (once per launch; tiny vs. total work)
__global__ void cvt_f32_f16(const float* __restrict__ s, _Float16* __restrict__ d, int n) {
    int i = blockIdx.x * blockDim.x + threadIdx.x;
    if (i < n) d[i] = (_Float16)s[i];
}

// Persistent fused LSTM layer:
//   gates[32, 32cols] = [x_t ; h] (32x1024, f16)  x  [W_ih ; W_hh]^T slice (1024x32, f16 in LDS)
// computed with v_wmma_f32_16x16x32_f16 (dual accumulators), f32 cell state in LDS.
// Activation staging uses CDNA5 GLOBAL_LOAD_ASYNC_TO_LDS_B128 (ASYNCcnt) where source is f16.
template<bool IN_HALF, bool OUT_F32>
__global__ __launch_bounds__(TPB)
void lstm_scan(const void* __restrict__ xin,          // [B,S,F] f32 (layer0) or f16 (layer1)
               const _Float16* __restrict__ wih,      // [4H, F] f16
               const _Float16* __restrict__ whh,      // [4H, H] f16
               const float* __restrict__ bih,         // [4H]
               const float* __restrict__ bhh,         // [4H]
               _Float16* __restrict__ hbuf,           // [B, H] f16, global recurrent state (pre-zeroed)
               void* __restrict__ yout,               // [B,S,H] f16 (layer0) or f32 (layer1)
               float* __restrict__ hT,                // [B,H] final h for this layer
               float* __restrict__ cT,                // [B,H] final c for this layer
               unsigned* __restrict__ bar)            // grid barrier counter (pre-zeroed)
{
    __shared__ _Float16 alds[BB * LP];        // [x_t ; h] staged per step
    __shared__ _Float16 wlds[NL * LP];        // weight slice, resident whole scan
    __shared__ float    gbuf[2][2][16][16];   // [mt][nt] 16x16 f32 gate tiles
    __shared__ float    clds[BB][HB];         // f32 cell state
    __shared__ float    biasl[NL];            // b_ih + b_hh for our columns

    const int tid = threadIdx.x;
    const int blk = blockIdx.x;

    // ---- one-time: load weight slice into LDS (columns = 4 gates x HB units) ----
    for (int idx = tid; idx < NL * KK; idx += TPB) {
        int n_local = idx >> 10;              // idx / KK
        int k       = idx & (KK - 1);
        int gate    = n_local >> 3;
        int u       = n_local & 7;
        int ng      = gate * HH + blk * HB + u;   // global gate row in [0,4H)
        _Float16 v  = (k < FF) ? wih[(size_t)ng * FF + k]
                               : whh[(size_t)ng * HH + (k - FF)];
        wlds[n_local * LP + k] = v;
    }
    if (tid < NL) {
        int gate = tid >> 3, u = tid & 7;
        int ng = gate * HH + blk * HB + u;
        biasl[tid] = bih[ng] + bhh[ng];
    }
    for (int p = tid; p < BB * HB; p += TPB) clds[p >> 3][p & 7] = 0.0f;
    __syncthreads();

    const int wave = tid >> 5;
    const int lane = tid & 31;
    const int mt   = wave >> 1;     // M-tile 0..1 (rows 0..15 / 16..31)
    const int nt   = wave & 1;      // N-tile 0..1 (cols 0..15 / 16..31 of local slice)
    const int r    = lane & 15;
    const int half = lane >> 4;

    for (int t = 0; t < SS; ++t) {
        // ---- stage x_t (first FF columns of alds) ----
        if (IN_HALF) {
            // f16 source: CDNA5 async copy straight into LDS (tracked by ASYNCcnt)
            const char* xb = (const char*)xin;
            #pragma unroll 4
            for (int i = 0; i < (BB * FF) / (TPB * 8); ++i) {     // 16 iters, 8 f16 per lane
                int elem = (i * TPB + tid) * 8;
                int m = elem >> 9;                                 // / FF
                int k = elem & (FF - 1);
                unsigned go = (unsigned)((size_t)(m * SS + t) * FF * 2 + (size_t)k * 2);
                unsigned lb = (unsigned)(size_t)&alds[m * LP + k];
                asm volatile("global_load_async_to_lds_b128 %0, %1, %2"
                             :: "v"(lb), "v"(go), "s"(xb) : "memory");
            }
        } else {
            // f32 source: convert on the fly through VALU
            for (int idx = tid; idx < BB * FF; idx += TPB) {
                int m = idx >> 9;
                int k = idx & (FF - 1);
                alds[m * LP + k] =
                    (_Float16)(((const float*)xin)[(size_t)(m * SS + t) * FF + k]);
            }
        }
        // ---- stage h (last HH columns of alds): async copy from global f16 state ----
        {
            const char* hb8 = (const char*)hbuf;
            #pragma unroll 4
            for (int i = 0; i < (BB * HH) / (TPB * 8); ++i) {      // 16 iters
                int elem = (i * TPB + tid) * 8;
                int m = elem >> 9;                                  // / HH
                int k = elem & (HH - 1);
                unsigned go = (unsigned)(elem * 2);
                unsigned lb = (unsigned)(size_t)&alds[m * LP + FF + k];
                asm volatile("global_load_async_to_lds_b128 %0, %1, %2"
                             :: "v"(lb), "v"(go), "s"(hb8) : "memory");
            }
        }
        asm volatile("s_wait_asynccnt 0x0" ::: "memory");
        __syncthreads();

        // ---- WMMA: one 16x16 f32 tile per wave, K = 1024, dual accumulators ----
        v8f acc0 = {}, acc1 = {};
        const _Float16* arow = &alds[(mt * 16 + r) * LP];   // A: lane r -> matrix row
        const _Float16* bcol = &wlds[(nt * 16 + r) * LP];   // B: lane r -> matrix col
        #pragma unroll 2
        for (int kc = 0; kc < KK; kc += 64) {
            v16h a0, b0, a1, b1;
            // A 16x32 f16, contiguous k order used here (B matches): lanes 0-15 hold K=0..15, lanes 16-31 hold K=16..31
            int ka0 = kc + half * 16;
            int ka1 = ka0 + 32;
            #pragma unroll
            for (int e = 0; e < 16; ++e) {
                a0[e] = arow[ka0 + e];
                a1[e] = arow[ka1 + e];
            }
            // B 32x16 f16, same contiguous k order as A: lanes 0-15 hold K=0..15, lanes 16-31 hold K=16..31
            int kb0 = kc + half * 16;
            int kb1 = kb0 + 32;
            #pragma unroll
            for (int e = 0; e < 16; ++e) { b0[e] = bcol[kb0 + e]; b1[e] = bcol[kb1 + e]; }
            acc0 = __builtin_amdgcn_wmma_f32_16x16x32_f16(
                false, a0, false, b0, (short)0, acc0, false, false);
            acc1 = __builtin_amdgcn_wmma_f32_16x16x32_f16(
                false, a1, false, b1, (short)0, acc1, false, false);
        }
        // C/D layout: VGPR j -> row j + 8*half, col = lane&15
        #pragma unroll
        for (int j = 0; j < 8; ++j) gbuf[mt][nt][j + 8 * half][r] = acc0[j] + acc1[j];
        __syncthreads();

        // ---- pointwise LSTM cell (f32), update c/h, emit outputs ----
        for (int p = tid; p < BB * HB; p += TPB) {
            int m = p >> 3;
            int u = p & 7;
            int rr = m & 15, mm = m >> 4;
            float gi = gbuf[mm][0][rr][u]     + biasl[u];
            float gf = gbuf[mm][0][rr][8 + u] + biasl[8 + u];
            float gg = gbuf[mm][1][rr][u]     + biasl[16 + u];
            float go = gbuf[mm][1][rr][8 + u] + biasl[24 + u];
            float iv = sigmoidf_(gi);
            float fv = sigmoidf_(gf);
            float gv = tanhf(gg);
            float ov = sigmoidf_(go);
            float c  = fv * clds[m][u] + iv * gv;
            clds[m][u] = c;
            float h  = ov * tanhf(c);
            int hb_  = blk * HB + u;
            hbuf[m * HH + hb_] = (_Float16)h;
            size_t yoff = (size_t)(m * SS + t) * HH + hb_;
            if (OUT_F32) ((float*)yout)[yoff]     = h;
            else         ((_Float16*)yout)[yoff]  = (_Float16)h;
            if (t == SS - 1) { hT[m * HH + hb_] = h; cT[m * HH + hb_] = c; }
        }

        // ---- monotonic grid barrier: all blocks finish step t before t+1 ----
        __syncthreads();
        if (tid == 0) {
            __threadfence();
            atomicAdd(bar, 1u);
            unsigned target = (unsigned)(NBLK * (t + 1));
            while (*(volatile unsigned*)bar < target) {
                __builtin_amdgcn_s_sleep(1);
            }
            __threadfence();
        }
        __syncthreads();
    }
}

extern "C" void kernel_launch(void* const* d_in, const int* in_sizes, int n_in,
                              void* d_out, int out_size, void* d_ws, size_t ws_size,
                              hipStream_t stream) {
    const float* x     = (const float*)d_in[0];
    const float* w_ih0 = (const float*)d_in[1];
    const float* w_hh0 = (const float*)d_in[2];
    const float* b_ih0 = (const float*)d_in[3];
    const float* b_hh0 = (const float*)d_in[4];
    const float* w_ih1 = (const float*)d_in[5];
    const float* w_hh1 = (const float*)d_in[6];
    const float* b_ih1 = (const float*)d_in[7];
    const float* b_hh1 = (const float*)d_in[8];

    const size_t W  = (size_t)4 * HH * FF;        // 2048*512 elements per weight matrix
    char* ws = (char*)d_ws;
    _Float16* wih0h = (_Float16*)(ws);                       // 2 MB each
    _Float16* whh0h = wih0h + W;
    _Float16* wih1h = whh0h + W;
    _Float16* whh1h = wih1h + W;
    _Float16* hbuf  = whh1h + W;                             // 32 KB
    unsigned* bar   = (unsigned*)((char*)(hbuf + (size_t)BB * HH)); // 256 B slot
    _Float16* y0f16 = (_Float16*)((char*)bar + 256);         // 64 MB [B,S,H] f16

    float* y1 = (float*)d_out;                               // [B,S,H] f32
    float* hT = y1 + (size_t)BB * SS * HH;                   // [2,B,H]
    float* cT = hT + (size_t)2 * BB * HH;                    // [2,B,H]

    // Convert weights to f16 once per launch.
    const int n = (int)W;
    cvt_f32_f16<<<(n + 255) / 256, 256, 0, stream>>>(w_ih0, wih0h, n);
    cvt_f32_f16<<<(n + 255) / 256, 256, 0, stream>>>(w_hh0, whh0h, n);
    cvt_f32_f16<<<(n + 255) / 256, 256, 0, stream>>>(w_ih1, wih1h, n);
    cvt_f32_f16<<<(n + 255) / 256, 256, 0, stream>>>(w_hh1, whh1h, n);

    // Layer 0: f32 input x -> y0 (f16 scratch)
    hipMemsetAsync(hbuf, 0, (size_t)BB * HH * sizeof(_Float16), stream);
    hipMemsetAsync(bar, 0, 256, stream);
    lstm_scan<false, false><<<NBLK, TPB, 0, stream>>>(
        (const void*)x, wih0h, whh0h, b_ih0, b_hh0, hbuf,
        (void*)y0f16, hT + 0, cT + 0, bar);

    // Layer 1: f16 input y0 -> y1 (f32 final output)
    hipMemsetAsync(hbuf, 0, (size_t)BB * HH * sizeof(_Float16), stream);
    hipMemsetAsync(bar, 0, 256, stream);
    lstm_scan<true, true><<<NBLK, TPB, 0, stream>>>(
        (const void*)y0f16, wih1h, whh1h, b_ih1, b_hh1, hbuf,
        (void*)y1, hT + (size_t)BB * HH, cT + (size_t)BB * HH, bar);
}
